// RouterKT_66434554135147
// MI455X (gfx1250) — compile-verified, matrix operand order checked by reading, not run on hardware
//
#include <hip/hip_runtime.h>
#include <hip/hip_bf16.h>
#include <stdint.h>

// ---------------- model constants (match reference) ----------------
#define BSZ      64
#define SEQL     512
#define DMODEL   256
#define NHEADS   8
#define DKH      32
#define HSHARED  1
#define HDYN     7
#define DFFN     1024
#define MTOK     (BSZ*SEQL)       // 32768 tokens
#define NEGINF   (-1e9f)

typedef __attribute__((ext_vector_type(16))) __bf16 bf16x16;
typedef __attribute__((ext_vector_type(8)))  float  floatx8;

union FragAB { bf16x16 v; unsigned int u[8]; uint4 q[2]; };

#if defined(__AMDGCN__) && __has_builtin(__builtin_amdgcn_global_load_async_to_lds_b128)
  #define HAVE_ASYNC_LDS 1
#else
  #define HAVE_ASYNC_LDS 0
#endif

#if HAVE_ASYNC_LDS
typedef int v4i_t __attribute__((ext_vector_type(4)));
typedef __attribute__((address_space(1))) v4i_t g_v4i;   // global (prints as __device__)
typedef __attribute__((address_space(3))) v4i_t l_v4i;   // LDS
#endif

// stage 16 bytes global -> LDS (async DMA on CDNA5; reg copy fallback)
__device__ __forceinline__ void stage_b128(unsigned short* lds, const unsigned short* g) {
#if HAVE_ASYNC_LDS
  __builtin_amdgcn_global_load_async_to_lds_b128((g_v4i*)g, (l_v4i*)lds, 0, 0);
#else
  *(uint4*)lds = *(const uint4*)g;
#endif
}

__device__ __forceinline__ void wait_async0() {
#if HAVE_ASYNC_LDS
#if __has_builtin(__builtin_amdgcn_s_wait_asynccnt)
  __builtin_amdgcn_s_wait_asynccnt(0);
#else
  asm volatile("s_wait_asynccnt 0x0" ::: "memory");
#endif
#endif
}

__device__ __forceinline__ void stage_fence() {
  wait_async0();       // my wave's async LDS writes landed
  __syncthreads();     // everyone's landed; everyone done reading the other buffer
}

__device__ __forceinline__ unsigned short f2bf(float f) {
  unsigned int u = __float_as_uint(f);
  unsigned int r = u + 0x7FFFu + ((u >> 16) & 1u);   // round-to-nearest-even
  return (unsigned short)(r >> 16);
}

// 16-bit WMMA A/B fragment (wave32): lane m/n = L&15, kbase = (L>=16)?8:0.
// u[0..3] = 8 contiguous bf16 at element offset kb, u[4..7] at 16+kb.
// => a fragment is exactly two b128 loads from a K-contiguous row.

// ---------------- generic WMMA GEMM: C[M,N] = A[M,K](bf16) @ W[N,K]^T + bias
// block = 256 threads = 8 waves, tile 128(M) x 64(N), K-step 32, double-buffered
// async global->LDS staging. wave (wm,wn) computes a 32x32 sub-tile (2x2 WMMA).
__global__ __launch_bounds__(256)
void gemm_bf16_wmma(const unsigned short* __restrict__ A,
                    const unsigned short* __restrict__ W,
                    const float* __restrict__ bias,
                    float* __restrict__ Cf,
                    unsigned short* __restrict__ Cb,
                    int K, int N, int relu)
{
  __shared__ __align__(16) unsigned short As[2][128*32];  // 2 x 8 KB
  __shared__ __align__(16) unsigned short Bs[2][64*32];   // 2 x 4 KB
  const int tid  = threadIdx.x;
  const int lane = tid & 31;
  const int wid  = tid >> 5;
  const int wm   = wid & 3;
  const int wn   = wid >> 2;
  const int m0   = blockIdx.x * 128;
  const int n0   = blockIdx.y * 64;

  // fixed per-thread staging coordinates (8 bf16 per chunk)
  const int eA0 = tid*8,        rA0 = eA0 >> 5, cA0 = eA0 & 31;
  const int eA1 = (tid+256)*8,  rA1 = eA1 >> 5, cA1 = eA1 & 31;
  const unsigned short* gA0 = A + (size_t)(m0+rA0)*K + cA0;
  const unsigned short* gA1 = A + (size_t)(m0+rA1)*K + cA1;
  const unsigned short* gB  = W + (size_t)(n0+rA0)*K + cA0;  // B tile: 64x32, chunk tid

  floatx8 acc[2][2] = {};
  int cur = 0;

  stage_b128(&As[0][eA0], gA0);
  stage_b128(&As[0][eA1], gA1);
  stage_b128(&Bs[0][eA0], gB);
  stage_fence();

  const int kb = ((lane >> 4) & 1) * 8;   // fragment K-base (elements)

  for (int k0 = 0; k0 < K; k0 += 32) {
    const int nxt = cur ^ 1;
    if (k0 + 32 < K) {                    // stage next tile while computing
      stage_b128(&As[nxt][eA0], gA0 + k0 + 32);
      stage_b128(&As[nxt][eA1], gA1 + k0 + 32);
      stage_b128(&Bs[nxt][eA0], gB  + k0 + 32);
    }

    FragAB af[2], bfr[2];
    #pragma unroll
    for (int i = 0; i < 2; ++i) {
      int row = wm*32 + i*16 + (lane & 15);
      af[i].q[0] = *(const uint4*)&As[cur][row*32 + kb];
      af[i].q[1] = *(const uint4*)&As[cur][row*32 + 16 + kb];
    }
    #pragma unroll
    for (int i = 0; i < 2; ++i) {
      int row = wn*32 + i*16 + (lane & 15);
      bfr[i].q[0] = *(const uint4*)&Bs[cur][row*32 + kb];
      bfr[i].q[1] = *(const uint4*)&Bs[cur][row*32 + 16 + kb];
    }
    #pragma unroll
    for (int i = 0; i < 2; ++i)
      #pragma unroll
      for (int j = 0; j < 2; ++j)
        acc[i][j] = __builtin_amdgcn_wmma_f32_16x16x32_bf16(
            false, af[i].v, false, bfr[j].v, (short)0, acc[i][j], false, false);

    if (k0 + 32 < K) stage_fence();
    cur = nxt;
  }

  // ---- epilogue: C layout lane: n=L&15, m = 8*(L>=16)+vgpr ----
  #pragma unroll
  for (int i = 0; i < 2; ++i) {
    #pragma unroll
    for (int j = 0; j < 2; ++j) {
      int n  = n0 + wn*32 + j*16 + (lane & 15);
      int mb = m0 + wm*32 + i*16 + ((lane >> 4) & 1)*8;
      float bv = bias ? bias[n] : 0.f;
      #pragma unroll
      for (int r = 0; r < 8; ++r) {
        float v = acc[i][j][r] + bv;
        if (relu) v = fmaxf(v, 0.f);
        size_t off = (size_t)(mb + r)*N + n;
        if (Cf) Cf[off] = v;
        if (Cb) Cb[off] = f2bf(v);
      }
    }
  }
}

// ---------------- fused attention: one block per (b*h, 16-row q block) ----
// scores = (Qh Qh^T)/sqrt(dk) masked -> softmax -> ctx = P Vh * routing[b,h]
__global__ __launch_bounds__(256)
void attn_fused(const unsigned short* __restrict__ Qh,  // [BSZ*NHEADS, SEQL, DKH] bf16
                const unsigned short* __restrict__ Vh,  // same layout
                const float* __restrict__ routing,      // [BSZ*NHEADS]
                unsigned short* __restrict__ ctx,       // [MTOK, DMODEL] bf16
                int mask_flag)                          // 1: key<=q; 0: key<q + zero_pad row0
{
  // 48 KB: scores f32 (phases 1-2) aliased by V^T bf16 (phase 3), then P bf16
  __shared__ __align__(16) unsigned char smem[16*SEQL*4 + 16*SEQL*2];
  __shared__ float sCtx[16*DKH];
  float*          sS  = (float*)smem;                        // 16 x 512 f32
  unsigned short* sVt = (unsigned short*)smem;               // 32 x 512 bf16 (alias)
  unsigned short* sP  = (unsigned short*)(smem + 16*SEQL*4); // 16 x 512 bf16

  const int tid = threadIdx.x, lane = tid & 31, wid = tid >> 5;
  const int bh = blockIdx.x, b = bh / NHEADS, h = bh % NHEADS;
  const int q0 = blockIdx.y * 16;
  const unsigned short* Qb = Qh + (size_t)bh * SEQL * DKH;
  const unsigned short* Vb = Vh + (size_t)bh * SEQL * DKH;
  const int kb = ((lane >> 4) & 1) * 8;

  for (int i = tid; i < 16*DKH; i += 256) sCtx[i] = 0.f;

  // ---- phase 1: scores (K = DKH = 32 -> single WMMA per 16x16 tile) ----
  FragAB aq;
  {
    int qr = q0 + (lane & 15);
    aq.q[0] = *(const uint4*)&Qb[qr*DKH + kb];
    aq.q[1] = *(const uint4*)&Qb[qr*DKH + 16 + kb];
  }
  const float scal = 0.17677669529663687f;  // 1/sqrt(32)
  #pragma unroll
  for (int t = 0; t < 4; ++t) {
    int kc   = (wid*4 + t)*16;
    int key  = kc + (lane & 15);
    int qrow = q0 + ((lane >> 4) & 1)*8;
    bool fully_masked = mask_flag ? (kc > q0 + 15) : (kc >= q0 + 15);
    if (fully_masked) {
      #pragma unroll
      for (int r = 0; r < 8; ++r) {
        int q = qrow + r;
        sS[(q - q0)*SEQL + key] = (!mask_flag && q == 0) ? 0.f : NEGINF;
      }
    } else {
      FragAB bk;
      bk.q[0] = *(const uint4*)&Qb[key*DKH + kb];
      bk.q[1] = *(const uint4*)&Qb[key*DKH + 16 + kb];
      floatx8 acc = {};
      acc = __builtin_amdgcn_wmma_f32_16x16x32_bf16(
          false, aq.v, false, bk.v, (short)0, acc, false, false);
      #pragma unroll
      for (int r = 0; r < 8; ++r) {
        int q = qrow + r;
        bool allowed = mask_flag ? (key <= q) : (key < q);
        float s = allowed ? acc[r]*scal : NEGINF;
        if (!mask_flag && q == 0) s = 0.f;   // zero_pad (pre-softmax)
        sS[(q - q0)*SEQL + key] = s;
      }
    }
  }
  __syncthreads();

  // ---- phase 2: row softmax (one wave per 2 rows), P -> bf16 in sP ----
  #pragma unroll
  for (int rr = 0; rr < 2; ++rr) {
    int row = wid*2 + rr;
    float mx = -3.4e38f;
    for (int c = lane; c < SEQL; c += 32) mx = fmaxf(mx, sS[row*SEQL + c]);
    #pragma unroll
    for (int off = 16; off > 0; off >>= 1) mx = fmaxf(mx, __shfl_xor(mx, off, 32));
    float sum = 0.f;
    for (int c = lane; c < SEQL; c += 32) {
      float e = __expf(sS[row*SEQL + c] - mx);
      sS[row*SEQL + c] = e;
      sum += e;
    }
    #pragma unroll
    for (int off = 16; off > 0; off >>= 1) sum += __shfl_xor(sum, off, 32);
    float inv = 1.f / sum;
    for (int c = lane; c < SEQL; c += 32)
      sP[row*SEQL + c] = f2bf(sS[row*SEQL + c] * inv);
  }
  __syncthreads();   // sS dead from here; reuse as sVt

  // ---- stage V^T into LDS: sVt[d][s] (vectorized b128 reads of V rows) ----
  for (int i = tid; i < (SEQL*DKH)/8; i += 256) {
    int s  = i >> 2;           // 4 chunks of 8 d-values per s-row
    int dc = (i & 3) * 8;
    uint4 tv = *(const uint4*)&Vb[(size_t)s*DKH + dc];
    const unsigned short* pv = (const unsigned short*)&tv;
    #pragma unroll
    for (int e = 0; e < 8; ++e) sVt[(dc + e)*SEQL + s] = pv[e];
  }
  __syncthreads();

  // ---- phase 3: ctx = P @ V  (16x32 output, K=512 split over 8 waves) ----
  floatx8 co[2] = {};
  #pragma unroll
  for (int ks = 0; ks < 2; ++ks) {
    int k0 = (wid*2 + ks)*32;
    FragAB ap;
    ap.q[0] = *(const uint4*)&sP[(lane & 15)*SEQL + k0 + kb];
    ap.q[1] = *(const uint4*)&sP[(lane & 15)*SEQL + k0 + 16 + kb];
    #pragma unroll
    for (int nt = 0; nt < 2; ++nt) {
      FragAB bvf;
      int d = nt*16 + (lane & 15);
      bvf.q[0] = *(const uint4*)&sVt[d*SEQL + k0 + kb];
      bvf.q[1] = *(const uint4*)&sVt[d*SEQL + k0 + 16 + kb];
      co[nt] = __builtin_amdgcn_wmma_f32_16x16x32_bf16(
          false, ap.v, false, bvf.v, (short)0, co[nt], false, false);
    }
  }
  #pragma unroll
  for (int nt = 0; nt < 2; ++nt) {
    int d  = nt*16 + (lane & 15);
    int qb = ((lane >> 4) & 1)*8;
    #pragma unroll
    for (int r = 0; r < 8; ++r)
      atomicAdd(&sCtx[(qb + r)*DKH + d], co[nt][r]);
  }
  __syncthreads();

  // ---- phase 4: scale by routing and write bf16 [b,s, h*dk+d] ----
  float rscale = routing[bh];
  for (int i = tid; i < 16*DKH; i += 256) {
    int q = i >> 5, d = i & 31;
    ctx[((size_t)(b*SEQL) + q0 + q)*DMODEL + h*DKH + d] = f2bf(sCtx[i] * rscale);
  }
}

// ---------------- routing: gates = softmax(q @ Wg^T), top-2 mask, mean over S
__global__ void routing_init(float* routing) {
  int i = threadIdx.x;
  if (i < BSZ*NHEADS) routing[i] = ((i % NHEADS) < HSHARED) ? 1.f : 0.f;
}

__global__ __launch_bounds__(256)
void routing_gates(const float* __restrict__ q, const float* __restrict__ Wg,
                   float* __restrict__ routing)
{
  __shared__ float sW[HDYN*DMODEL];
  for (int i = threadIdx.x; i < HDYN*DMODEL; i += 256) sW[i] = Wg[i];
  __syncthreads();
  int m = blockIdx.x*256 + threadIdx.x;
  int b = m / SEQL;
  float g[HDYN];
  #pragma unroll
  for (int hh = 0; hh < HDYN; ++hh) {
    float acc = 0.f;
    for (int k = 0; k < DMODEL; ++k) acc += q[(size_t)m*DMODEL + k] * sW[hh*DMODEL + k];
    g[hh] = acc;
  }
  float mx = g[0];
  #pragma unroll
  for (int hh = 1; hh < HDYN; ++hh) mx = fmaxf(mx, g[hh]);
  float sum = 0.f;
  #pragma unroll
  for (int hh = 0; hh < HDYN; ++hh) { g[hh] = __expf(g[hh] - mx); sum += g[hh]; }
  float inv = 1.f / sum;
  #pragma unroll
  for (int hh = 0; hh < HDYN; ++hh) g[hh] *= inv;
  int i1 = 0;
  #pragma unroll
  for (int hh = 1; hh < HDYN; ++hh) if (g[hh] > g[i1]) i1 = hh;
  int i2 = (i1 == 0) ? 1 : 0;
  #pragma unroll
  for (int hh = 0; hh < HDYN; ++hh) if (hh != i1 && g[hh] > g[i2]) i2 = hh;
  const float invS = 1.f / SEQL;
  #pragma unroll
  for (int hh = 0; hh < HDYN; ++hh) {
    float val = (hh == i1 || hh == i2) ? g[hh] : 0.f;
    if (val != 0.f) atomicAdd(&routing[b*NHEADS + HSHARED + hh], val * invS);
  }
}

// ---------------- residual + LayerNorm (row = one block of 256 = DMODEL) ----
__global__ __launch_bounds__(256)
void add_ln(float* __restrict__ x, const float* __restrict__ r,
            const float* __restrict__ g, const float* __restrict__ b,
            unsigned short* __restrict__ xbf)
{
  __shared__ float red[8];
  int row = blockIdx.x, i = threadIdx.x;
  float v = x[(size_t)row*DMODEL + i] + r[(size_t)row*DMODEL + i];
  float s = v;
  #pragma unroll
  for (int off = 16; off > 0; off >>= 1) s += __shfl_xor(s, off, 32);
  if ((i & 31) == 0) red[i >> 5] = s;
  __syncthreads();
  float tot = 0.f;
  #pragma unroll
  for (int w = 0; w < 8; ++w) tot += red[w];
  float mean = tot * (1.f / DMODEL);
  __syncthreads();
  float d = v - mean, s2 = d*d;
  #pragma unroll
  for (int off = 16; off > 0; off >>= 1) s2 += __shfl_xor(s2, off, 32);
  if ((i & 31) == 0) red[i >> 5] = s2;
  __syncthreads();
  float tot2 = 0.f;
  #pragma unroll
  for (int w = 0; w < 8; ++w) tot2 += red[w];
  float var = tot2 * (1.f / DMODEL);
  float out = d * rsqrtf(var + 1e-5f) * g[i] + b[i];
  x[(size_t)row*DMODEL + i]   = out;
  xbf[(size_t)row*DMODEL + i] = f2bf(out);
}

// ---------------- small helpers ----------------
__global__ void repack_heads(const float* __restrict__ in, unsigned short* __restrict__ out) {
  size_t idx = (size_t)blockIdx.x*256 + threadIdx.x;  // [b,s,h,d] -> [b,h,s,d] bf16
  int d = idx & (DKH-1);
  size_t t = idx >> 5;
  int hh = t & (NHEADS-1);
  size_t bs = t >> 3;
  int s = bs & (SEQL-1);
  int b = (int)(bs >> 9);
  out[(((size_t)(b*NHEADS + hh)*SEQL) + s)*DKH + d] = f2bf(in[idx]);
}

__global__ void cvt_bf16_k(const float* __restrict__ in, unsigned short* __restrict__ out, size_t n) {
  size_t i = (size_t)blockIdx.x*256 + threadIdx.x;
  if (i < n) out[i] = f2bf(in[i]);
}
__global__ void copy_f32_k(const float* __restrict__ in, float* __restrict__ out, size_t n) {
  size_t i = (size_t)blockIdx.x*256 + threadIdx.x;
  if (i < n) out[i] = in[i];
}

// ---------------- host orchestration ----------------
extern "C" void kernel_launch(void* const* d_in, const int* in_sizes, int n_in,
                              void* d_out, int out_size, void* d_ws, size_t ws_size,
                              hipStream_t stream) {
  (void)in_sizes; (void)n_in; (void)out_size; (void)ws_size;
  const float* q_embed  = (const float*)d_in[0];
  const float* qa_embed = (const float*)d_in[1];
  // dict order per layer set: Wq,bq,Wv,bv,Wg,Wo,bo,g1,b1,W1,b1f,W2,b2f,g2,b2
  const float* P1[15]; const float* P2[15];
  for (int i = 0; i < 15; ++i) { P1[i] = (const float*)d_in[2+i]; P2[i] = (const float*)d_in[17+i]; }

  char* ws = (char*)d_ws;
  size_t off = 0;
  auto alloc = [&](size_t bytes) -> char* {
    off = (off + 255) & ~(size_t)255;
    char* p = ws + off; off += bytes; return p;
  };

  // bf16 weights (6 layers total, ~8.3 MB)
  unsigned short* wq1 = (unsigned short*)alloc((size_t)2*DMODEL*DMODEL*2);
  unsigned short* wv1 = (unsigned short*)alloc((size_t)2*DMODEL*DMODEL*2);
  unsigned short* wo1 = (unsigned short*)alloc((size_t)2*DMODEL*DMODEL*2);
  unsigned short* w11 = (unsigned short*)alloc((size_t)2*DFFN*DMODEL*2);
  unsigned short* w21 = (unsigned short*)alloc((size_t)2*DMODEL*DFFN*2);
  unsigned short* wq2 = (unsigned short*)alloc((size_t)4*DMODEL*DMODEL*2);
  unsigned short* wv2 = (unsigned short*)alloc((size_t)4*DMODEL*DMODEL*2);
  unsigned short* wo2 = (unsigned short*)alloc((size_t)4*DMODEL*DMODEL*2);
  unsigned short* w12 = (unsigned short*)alloc((size_t)4*DFFN*DMODEL*2);
  unsigned short* w22 = (unsigned short*)alloc((size_t)4*DMODEL*DFFN*2);
  // activations / scratch (~310 MB)
  const size_t nact = (size_t)MTOK*DMODEL;
  float*          x_f32 = (float*)alloc(nact*4);
  unsigned short* x_bf  = (unsigned short*)alloc(nact*2);
  float*          y_f32 = (float*)alloc(nact*4);
  unsigned short* y_bf  = (unsigned short*)alloc(nact*2);
  float*          q_f32 = (float*)alloc(nact*4);
  float*          v_f32 = (float*)alloc(nact*4);
  float*          o_f32 = (float*)alloc(nact*4);
  unsigned short* qh_bf = (unsigned short*)alloc(nact*2);
  unsigned short* vh_bf = (unsigned short*)alloc(nact*2);
  unsigned short* ctx_bf= (unsigned short*)alloc(nact*2);
  unsigned short* h1_bf = (unsigned short*)alloc((size_t)MTOK*DFFN*2);
  float*          routing = (float*)alloc((size_t)BSZ*NHEADS*4);

  auto cvt = [&](const float* in, unsigned short* out, size_t n) {
    cvt_bf16_k<<<dim3((unsigned)((n + 255)/256)), dim3(256), 0, stream>>>(in, out, n);
  };
  auto gemm = [&](const unsigned short* A, const unsigned short* W, const float* bias,
                  float* Cf, unsigned short* Cb, int K, int N, int relu) {
    gemm_bf16_wmma<<<dim3(MTOK/128, N/64), dim3(256), 0, stream>>>(A, W, bias, Cf, Cb, K, N, relu);
  };

  // convert all weights once per launch (deterministic, no caching)
  cvt(P1[0],  wq1, (size_t)2*DMODEL*DMODEL);
  cvt(P1[2],  wv1, (size_t)2*DMODEL*DMODEL);
  cvt(P1[5],  wo1, (size_t)2*DMODEL*DMODEL);
  cvt(P1[9],  w11, (size_t)2*DFFN*DMODEL);
  cvt(P1[11], w21, (size_t)2*DMODEL*DFFN);
  cvt(P2[0],  wq2, (size_t)4*DMODEL*DMODEL);
  cvt(P2[2],  wv2, (size_t)4*DMODEL*DMODEL);
  cvt(P2[5],  wo2, (size_t)4*DMODEL*DMODEL);
  cvt(P2[9],  w12, (size_t)4*DFFN*DMODEL);
  cvt(P2[11], w22, (size_t)4*DMODEL*DFFN);

  auto tlayer = [&](const float* const* P,
                    const unsigned short* wq, const unsigned short* wv,
                    const unsigned short* wo, const unsigned short* w1,
                    const unsigned short* w2, int layer,
                    float* act_f, unsigned short* act_b,
                    const unsigned short* val_b,
                    int mask_flag, int apply_pos)
  {
    const float* bq  = P[1]  + (size_t)layer*DMODEL;
    const float* bv  = P[3]  + (size_t)layer*DMODEL;
    const float* Wg  = P[4]  + (size_t)layer*HDYN*DMODEL;
    const float* bo  = P[6]  + (size_t)layer*DMODEL;
    const float* g1  = P[7]  + (size_t)layer*DMODEL;
    const float* b1  = P[8]  + (size_t)layer*DMODEL;
    const float* b1f = P[10] + (size_t)layer*DFFN;
    const float* b2f = P[12] + (size_t)layer*DMODEL;
    const float* g2  = P[13] + (size_t)layer*DMODEL;
    const float* b2  = P[14] + (size_t)layer*DMODEL;
    const unsigned short* wqL = wq + (size_t)layer*DMODEL*DMODEL;
    const unsigned short* wvL = wv + (size_t)layer*DMODEL*DMODEL;
    const unsigned short* woL = wo + (size_t)layer*DMODEL*DMODEL;
    const unsigned short* w1L = w1 + (size_t)layer*DFFN*DMODEL;
    const unsigned short* w2L = w2 + (size_t)layer*DMODEL*DFFN;

    gemm(act_b, wqL, bq, q_f32, nullptr, DMODEL, DMODEL, 0);
    gemm(val_b, wvL, bv, v_f32, nullptr, DMODEL, DMODEL, 0);
    repack_heads<<<dim3((unsigned)(nact/256)), dim3(256), 0, stream>>>(q_f32, qh_bf);
    repack_heads<<<dim3((unsigned)(nact/256)), dim3(256), 0, stream>>>(v_f32, vh_bf);
    routing_init<<<dim3(1), dim3(512), 0, stream>>>(routing);
    routing_gates<<<dim3(MTOK/256), dim3(256), 0, stream>>>(q_f32, Wg, routing);
    attn_fused<<<dim3(BSZ*NHEADS, SEQL/16), dim3(256), 0, stream>>>(
        qh_bf, vh_bf, routing, ctx_bf, mask_flag);
    gemm(ctx_bf, woL, bo, o_f32, nullptr, DMODEL, DMODEL, 0);
    add_ln<<<dim3(MTOK), dim3(256), 0, stream>>>(act_f, o_f32, g1, b1, act_b);
    if (apply_pos) {
      gemm(act_b, w1L, b1f, nullptr, h1_bf, DMODEL, DFFN, 1);
      gemm(h1_bf, w2L, b2f, o_f32, nullptr, DFFN, DMODEL, 0);
      add_ln<<<dim3(MTOK), dim3(256), 0, stream>>>(act_f, o_f32, g2, b2, act_b);
    }
  };

  // init activations (never mutate d_in)
  copy_f32_k<<<dim3((unsigned)((nact+255)/256)), dim3(256), 0, stream>>>(q_embed,  x_f32, nact);
  cvt(q_embed,  x_bf, nact);
  copy_f32_k<<<dim3((unsigned)((nact+255)/256)), dim3(256), 0, stream>>>(qa_embed, y_f32, nact);
  cvt(qa_embed, y_bf, nact);

  // encoder stack on y (params1, 2 layers, self-attn, FFN)
  for (int i = 0; i < 2; ++i)
    tlayer(P1, wq1, wv1, wo1, w11, w21, i, y_f32, y_bf, y_bf, 1, 1);
  // decoder-ish stack on x (params2, 4 layers)
  for (int i = 0; i < 4; ++i) {
    if ((i & 1) == 0) tlayer(P2, wq2, wv2, wo2, w12, w22, i, x_f32, x_bf, x_bf, 1, 0);
    else              tlayer(P2, wq2, wv2, wo2, w12, w22, i, x_f32, x_bf, y_bf, 0, 1);
  }

  copy_f32_k<<<dim3((unsigned)((nact+255)/256)), dim3(256), 0, stream>>>(x_f32, (float*)d_out, nact);
}
